// Hypergraph_74672301408657
// MI455X (gfx1250) — matrix operand order repaired and checked by reading, not compiled, not run on hardware
//
#include <hip/hip_runtime.h>
#include <hip/hip_bf16.h>

#define N_NODES 50000
#define N_EDGES 50000
#define CIN 256
#define COUT 128
#define N_CAT 1000
#define N_AV  5000

typedef float  v2f   __attribute__((ext_vector_type(2)));
typedef float  v8f   __attribute__((ext_vector_type(8)));
typedef __bf16 v16bf __attribute__((ext_vector_type(16)));

// ---------------- utility kernels ----------------

__global__ void hg_zero_f32(float* __restrict__ p, int n) {
    int i = blockIdx.x * blockDim.x + threadIdx.x;
    int stride = gridDim.x * blockDim.x;
    for (; i < n; i += stride) p[i] = 0.0f;
}

__global__ void hg_degree(const int* __restrict__ node_i, const int* __restrict__ edge_i,
                          float* __restrict__ D, float* __restrict__ B, int E) {
    int e = blockIdx.x * blockDim.x + threadIdx.x;
    if (e >= E) return;
    atomicAdd(&D[node_i[e]], 1.0f);
    atomicAdd(&B[edge_i[e]], 1.0f);
}

__global__ void hg_invert(float* __restrict__ p, int n) {
    int i = blockIdx.x * blockDim.x + threadIdx.x;
    if (i >= n) return;
    float v = p[i];
    p[i] = (v > 0.0f) ? (1.0f / v) : 0.0f;
}

// ---------------- WMMA GEMM: x = product @ lin_w (fp32) ----------------
// One wave per 16x16 output tile; K=256 loop. N_NODES, CIN, COUT are all
// multiples of the tile dims, so EXEC stays all-ones for every WMMA.

__global__ void __launch_bounds__(128) hg_gemm_wmma(const float* __restrict__ A,   // [N_NODES, CIN]
                                                    const float* __restrict__ W,   // [CIN, COUT]
                                                    float* __restrict__ X) {       // [N_NODES, COUT]
    const int lane = threadIdx.x & 31;
    const int wave = threadIdx.x >> 5;
    const int tile = blockIdx.x * 4 + wave;
    const int TN = COUT / 16;                 // 8 tiles along N
    const int TM = N_NODES / 16;              // 3125 tiles along M
    if (tile >= TM * TN) return;              // uniform per wave -> no divergence
    const int tm = tile / TN;
    const int tn = tile % TN;
    const int h   = lane >> 4;                // lane group (0: lanes 0-15, 1: lanes 16-31)
    const int l16 = lane & 15;

    v8f acc = {0.0f, 0.0f, 0.0f, 0.0f, 0.0f, 0.0f, 0.0f, 0.0f};
    const float* Arow = A + (size_t)(tm * 16 + l16) * CIN;   // A row M = tm*16 + l16
    const float* Wcol = W + tn * 16 + l16;                   // B col N = tn*16 + l16

#if __has_builtin(__builtin_amdgcn_wmma_f32_16x16x4_f32)
    // fp32 WMMA: A 16x4 (vgpr v, group h -> K = v + 2h), B 4x16 (same K split),
    // C 16x16 (vgpr v, group h -> M = v + 8h).
    for (int k = 0; k < CIN; k += 4) {
        v2f a = *(const v2f*)(Arow + k + 2 * h);             // K = k+2h, k+2h+1
        v2f b;
        b.x = Wcol[(size_t)(k + 2 * h + 0) * COUT];
        b.y = Wcol[(size_t)(k + 2 * h + 1) * COUT];
        acc = __builtin_amdgcn_wmma_f32_16x16x4_f32(
            false, a, false, b, (short)0, acc, false, false);
    }
#else
    // bf16 fallback (codegen-verified builtin). A 16x32 k order chosen here (not AMD's table order):
    // element i, group h -> K = i + 16h. B 32x16: K = i + 16h.
    for (int k = 0; k < CIN; k += 32) {
        v16bf a, b;
#pragma unroll
        for (int i = 0; i < 16; ++i) {
            int ka = k + i + 16 * h;
            int kb = k + i + 16 * h;
            a[i] = (__bf16)Arow[ka];
            b[i] = (__bf16)Wcol[(size_t)kb * COUT];
        }
        acc = __builtin_amdgcn_wmma_f32_16x16x32_bf16(
            false, a, false, b, (short)0, acc, false, false);
    }
#endif

    float* Xout = X + tn * 16 + l16;
#pragma unroll
    for (int v = 0; v < 8; ++v) {
        Xout[(size_t)(tm * 16 + v + 8 * h) * COUT] = acc[v];
    }
}

// ---------------- scatter passes (one wave per incidence pair) ----------------
// 32 lanes x float4 = 128 channels per edge; all operands are L2-resident.

__global__ void hg_scatter_n2e(const int* __restrict__ node_i, const int* __restrict__ edge_i,
                               const float* __restrict__ x, float* __restrict__ ef, int E) {
    int e = blockIdx.x * 8 + (threadIdx.x >> 5);
    if (e >= E) return;
    int lane = threadIdx.x & 31;
    int n  = node_i[e];
    int ed = edge_i[e];
    float4 v = ((const float4*)(x + (size_t)n * COUT))[lane];
    float* dst = ef + (size_t)ed * COUT + lane * 4;
    atomicAdd(dst + 0, v.x);
    atomicAdd(dst + 1, v.y);
    atomicAdd(dst + 2, v.z);
    atomicAdd(dst + 3, v.w);
}

// edges -> nodes with B_inv, D_inv and view weight folded into one scalar,
// accumulating directly into the output buffer.
__global__ void hg_scatter_e2n(const int* __restrict__ node_i, const int* __restrict__ edge_i,
                               const float* __restrict__ ef, const float* __restrict__ Binv,
                               const float* __restrict__ Dinv, float* __restrict__ out,
                               int E, float w) {
    int e = blockIdx.x * 8 + (threadIdx.x >> 5);
    if (e >= E) return;
    int lane = threadIdx.x & 31;
    int n  = node_i[e];
    int ed = edge_i[e];
    float s = w * Binv[ed] * Dinv[n];
    float4 v = ((const float4*)(ef + (size_t)ed * COUT))[lane];
    float* dst = out + (size_t)n * COUT + lane * 4;
    atomicAdd(dst + 0, s * v.x);
    atomicAdd(dst + 1, s * v.y);
    atomicAdd(dst + 2, s * v.z);
    atomicAdd(dst + 3, s * v.w);
}

__global__ void hg_bias(float* __restrict__ out, const float* __restrict__ bias, int total) {
    int i = blockIdx.x * blockDim.x + threadIdx.x;
    if (i >= total) return;
    out[i] += bias[i & (COUT - 1)];
}

__global__ void hg_copy(float* __restrict__ dst, const float* __restrict__ src, int n) {
    int i = blockIdx.x * blockDim.x + threadIdx.x;
    if (i < n) dst[i] = src[i];
}

// ---------------- launch ----------------

extern "C" void kernel_launch(void* const* d_in, const int* in_sizes, int n_in,
                              void* d_out, int out_size, void* d_ws, size_t ws_size,
                              hipStream_t stream) {
    const float* product  = (const float*)d_in[0];
    const float* category = (const float*)d_in[1];
    const float* av       = (const float*)d_in[2];
    const int*   also_view = (const int*)d_in[3];
    const int*   also_buy  = (const int*)d_in[4];
    const float* lin_w    = (const float*)d_in[5];
    const float* bias     = (const float*)d_in[6];

    const int E = in_sizes[3] / 2;           // 1,600,000
    const int* av_node = also_view;          // row 0
    const int* av_edge = also_view + E;      // row 1
    const int* ab_node = also_buy;
    const int* ab_edge = also_buy + E;

    float* out     = (float*)d_out;
    float* out_cat = out + (size_t)N_NODES * COUT;
    float* out_av  = out_cat + (size_t)N_CAT * COUT;

    // workspace: x | edge_feat | degrees (Dv, Bv, Db, Bb)
    float* x   = (float*)d_ws;
    float* ef  = x + (size_t)N_NODES * COUT;
    float* deg = ef + (size_t)N_EDGES * COUT;
    float* Dv = deg;
    float* Bv = deg + N_NODES;
    float* Db = deg + 2 * N_NODES;
    float* Bb = deg + 3 * N_NODES;

    const int OUT_P = N_NODES * COUT;        // 6,400,000

    // degrees + output accumulator init
    hg_zero_f32<<<1024, 256, 0, stream>>>(deg, 4 * N_NODES);
    hg_zero_f32<<<4096, 256, 0, stream>>>(out, OUT_P);
    hg_degree<<<(E + 255) / 256, 256, 0, stream>>>(av_node, av_edge, Dv, Bv, E);
    hg_degree<<<(E + 255) / 256, 256, 0, stream>>>(ab_node, ab_edge, Db, Bb, E);
    hg_invert<<<(4 * N_NODES + 255) / 256, 256, 0, stream>>>(deg, 4 * N_NODES);

    // x = product @ lin_w (shared across views)
    const int tiles = (N_NODES / 16) * (COUT / 16);   // 25000
    hg_gemm_wmma<<<(tiles + 3) / 4, 128, 0, stream>>>(product, lin_w, x);

    // also_view pass
    hg_zero_f32<<<4096, 256, 0, stream>>>(ef, N_EDGES * COUT);
    hg_scatter_n2e<<<(E + 7) / 8, 256, 0, stream>>>(av_node, av_edge, x, ef, E);
    hg_scatter_e2n<<<(E + 7) / 8, 256, 0, stream>>>(av_node, av_edge, ef, Bv, Dv, out, E, 0.5f);

    // also_buy pass (reuses edge_feat buffer; stream order guarantees safety)
    hg_zero_f32<<<4096, 256, 0, stream>>>(ef, N_EDGES * COUT);
    hg_scatter_n2e<<<(E + 7) / 8, 256, 0, stream>>>(ab_node, ab_edge, x, ef, E);
    hg_scatter_e2n<<<(E + 7) / 8, 256, 0, stream>>>(ab_node, ab_edge, ef, Bb, Db, out, E, 0.5f);

    // bias (0.5*b + 0.5*b == b) and passthrough outputs
    hg_bias<<<(OUT_P + 255) / 256, 256, 0, stream>>>(out, bias, OUT_P);
    hg_copy<<<(N_CAT * COUT + 255) / 256, 256, 0, stream>>>(out_cat, category, N_CAT * COUT);
    hg_copy<<<(N_AV * COUT + 255) / 256, 256, 0, stream>>>(out_av, av, N_AV * COUT);
}
